// GeometricGAT_19327352832523
// MI455X (gfx1250) — compile-verified, matrix operand order checked by reading, not run on hardware
//
#include <hip/hip_runtime.h>

// ---------------------------------------------------------------------------
// GeometricGAT on MI455X (gfx1250, wave32).
// Dense node transforms -> v_wmma_f32_16x16x32_bf16, 16x64 strip per wave,
//   weights pre-packed to fragment order and staged to LDS with
//   global_load_async_to_lds_b128 (ASYNCcnt) for 8-wave reuse.
// Edge softmax/aggregation -> wave-per-edge, L2-resident gathers + atomics.
// ---------------------------------------------------------------------------

typedef __bf16 v16bf __attribute__((ext_vector_type(16)));
typedef float  v8f   __attribute__((ext_vector_type(8)));

#define GAT_N   20000
#define GAT_E   320000
#define GAT_IN  128
#define GAT_HID 256

// ---------------------------------------------------------------------------
// f32 -> bf16 conversion (activations for WMMA A-side)
// ---------------------------------------------------------------------------
__global__ void cvt_f32_bf16(const float* __restrict__ in,
                             __bf16* __restrict__ out, int n) {
  int i = blockIdx.x * blockDim.x + threadIdx.x;
  if (i < n) out[i] = (__bf16)in[i];
}

// ---------------------------------------------------------------------------
// Pack f32 weight [K,256] into bf16 WMMA-B fragment order:
//   Wp[ntile][kstep][lane][j]  (j = 0..15, 32 contiguous bytes per lane)
// Fragment layout per CDNA5 ISA 7.12.2 (16-bit B 32x16, lane = N column).
// ---------------------------------------------------------------------------
__global__ void pack_w_bf16(const float* __restrict__ W,
                            __bf16* __restrict__ Wp, int K) {
  const int t = blockIdx.x * blockDim.x + threadIdx.x;
  if (t >= K * 256) return;
  const int j     = t & 15;
  const int lane  = (t >> 4) & 31;
  const int rest  = t >> 9;
  const int kst   = (K >> 5);          // ksteps (power of 2)
  const int kstep = rest & (kst - 1);
  const int ntile = rest / kst;
  const int n  = ntile * 16 + (lane & 15);
  const int v  = j >> 1;
  const int kk = kstep * 32 + ((lane >> 4) & 1) * 8 +
                 ((v & 3) << 1) + ((v >> 2) << 4) + (j & 1);
  Wp[t] = (__bf16)W[kk * 256 + n];
}

// ---------------------------------------------------------------------------
// WMMA GEMM:  C[M,256] = A[M,K](bf16) @ W[K,256] + bias[256]
// block = 256 threads = 8 waves; grid = (4 ntile-groups, ceil(M/16/8)).
// Each wave owns a 16x64 output strip: 1 A fragment -> 4 WMMAs per K-step.
// Packed B slab (16..32KB) async-copied to LDS once per block.
// ---------------------------------------------------------------------------
__global__ __launch_bounds__(256)
void wmma_gemm_bf16(const __bf16* __restrict__ A,    // [M,K] row-major
                    const __bf16* __restrict__ Wp,   // packed fragment order
                    const float* __restrict__ bias,  // [256]
                    float* __restrict__ C,           // [M,256]
                    int M, int K) {
  constexpr int Nc = 256;
  __shared__ __align__(16) __bf16 sB[16384];         // up to 32KB (K=256)

  const int ksteps    = K >> 5;
  const int slabElems = 4 * ksteps * 512;            // this block's B slab
  const __bf16* __restrict__ gsrc = Wp + (size_t)blockIdx.x * slabElems;
  const int nchunks = slabElems >> 3;                // 16B chunks
  const unsigned sbase = (unsigned)(unsigned long long)(void*)&sB[0];

  // cooperative async stage: global -> LDS (ASYNCcnt)
  for (int i = threadIdx.x; i < nchunks; i += 256) {
    asm volatile("global_load_async_to_lds_b128 %0, %1, off"
                 :: "v"(sbase + i * 16), "v"(gsrc + (size_t)i * 8)
                 : "memory");
  }
  asm volatile("s_wait_asynccnt 0x0" ::: "memory");
  __syncthreads();

  const int lane  = threadIdx.x & 31;
  const int wave  = threadIdx.x >> 5;
  const int mtile = blockIdx.y * 8 + wave;
  if (mtile * 16 >= M) return;                       // wave-uniform guard

  const int m     = mtile * 16 + (lane & 15);
  const int kbase = ((lane >> 4) & 1) * 8;
  const __bf16* __restrict__ Arow = A + (size_t)m * K;

  v8f acc0 = {}, acc1 = {}, acc2 = {}, acc3 = {};
  for (int ks = 0; ks < ksteps; ++ks) {
    const int k0 = ks * 32;
    v16bf a;
#pragma unroll
    for (int v = 0; v < 8; ++v) {
      const int kk = k0 + kbase + ((v & 3) << 1) + ((v >> 2) << 4);
      a[2 * v]     = Arow[kk];
      a[2 * v + 1] = Arow[kk + 1];
    }
    if (k0 + 32 < K) __builtin_prefetch(&Arow[k0 + 32], 0, 0);

    v16bf b;
    const __bf16* sb;
#define GAT_LOADB(nt)                                                  \
    sb = &sB[(((nt) * ksteps + ks) * 32 + lane) * 16];                 \
    _Pragma("unroll") for (int j = 0; j < 16; ++j) b[j] = sb[j];

    GAT_LOADB(0)
    acc0 = __builtin_amdgcn_wmma_f32_16x16x32_bf16(false, a, false, b,
                                                   (short)0, acc0, false, false);
    GAT_LOADB(1)
    acc1 = __builtin_amdgcn_wmma_f32_16x16x32_bf16(false, a, false, b,
                                                   (short)0, acc1, false, false);
    GAT_LOADB(2)
    acc2 = __builtin_amdgcn_wmma_f32_16x16x32_bf16(false, a, false, b,
                                                   (short)0, acc2, false, false);
    GAT_LOADB(3)
    acc3 = __builtin_amdgcn_wmma_f32_16x16x32_bf16(false, a, false, b,
                                                   (short)0, acc3, false, false);
#undef GAT_LOADB
  }

  const int ncol = blockIdx.x * 64 + (lane & 15);
  const float bn0 = bias[ncol];
  const float bn1 = bias[ncol + 16];
  const float bn2 = bias[ncol + 32];
  const float bn3 = bias[ncol + 48];
#pragma unroll
  for (int r = 0; r < 8; ++r) {
    const int mm = mtile * 16 + r + ((lane >> 4) & 1) * 8;  // D: VGPR r -> M=r / r+8
    float* __restrict__ Crow = C + (size_t)mm * Nc + ncol;
    Crow[0]  = acc0[r] + bn0;
    Crow[16] = acc1[r] + bn1;
    Crow[32] = acc2[r] + bn2;
    Crow[48] = acc3[r] + bn3;
  }
}

// ---------------------------------------------------------------------------
// Per-layer reset: agg = 0, denom = 0, nmax = enc(-inf) = 0x007fffff
// ---------------------------------------------------------------------------
__global__ void init_layer(float* __restrict__ agg, float* __restrict__ denom,
                           unsigned* __restrict__ nmax) {
  int i = blockIdx.x * blockDim.x + threadIdx.x;
  if (i < GAT_N * 256) agg[i] = 0.f;
  if (i < GAT_N * 4) { denom[i] = 0.f; nmax[i] = 0x007fffffu; }
}

__device__ __forceinline__ unsigned enc_f32(float f) {
  unsigned u = __float_as_uint(f);
  return (u & 0x80000000u) ? ~u : (u | 0x80000000u);
}
__device__ __forceinline__ float dec_f32(unsigned u) {
  return __uint_as_float((u & 0x80000000u) ? (u & 0x7fffffffu) : ~u);
}

// ---------------------------------------------------------------------------
// Pass 1: per-edge GATv2 logits + segment-max (atomic umax on encoded f32).
// ---------------------------------------------------------------------------
__global__ void edge_logits(const int* __restrict__ ei,    // [2,E]
                            const float* __restrict__ eattr,
                            const float* __restrict__ xl,
                            const float* __restrict__ xr,
                            const float* __restrict__ we,   // [3,256]
                            const float* __restrict__ att,  // [256]
                            float* __restrict__ logits,     // [E,4]
                            unsigned* __restrict__ nmax) {
  const int lane = threadIdx.x & 31;
  const int e = blockIdx.x * (blockDim.x >> 5) + (threadIdx.x >> 5);
  if (e >= GAT_E) return;
  const int s = ei[e];
  const int d = ei[GAT_E + e];
  const float ea0 = eattr[e * 3 + 0];
  const float ea1 = eattr[e * 3 + 1];
  const float ea2 = eattr[e * 3 + 2];
  const float* __restrict__ xls = xl + (size_t)s * 256;
  const float* __restrict__ xrd = xr + (size_t)d * 256;

  float acc[4] = {0.f, 0.f, 0.f, 0.f};
#pragma unroll
  for (int it = 0; it < 8; ++it) {
    const int c = it * 32 + lane;
    float v = xls[c] + xrd[c] + ea0 * we[c] + ea1 * we[256 + c] + ea2 * we[512 + c];
    v = (v > 0.f) ? v : 0.2f * v;          // leaky_relu(0.2)
    acc[it >> 1] += v * att[c];
  }
#pragma unroll
  for (int h = 0; h < 4; ++h) {
    float v = acc[h];
#pragma unroll
    for (int off = 16; off > 0; off >>= 1) v += __shfl_xor(v, off, 32);
    if (lane == 0) {
      logits[e * 4 + h] = v;
      atomicMax(&nmax[d * 4 + h], enc_f32(v));
    }
  }
}

// ---------------------------------------------------------------------------
// Pass 2: p = exp(logit - max[dst]); denom[dst] += p   (thread per (e,h))
// ---------------------------------------------------------------------------
__global__ void edge_expsum(const int* __restrict__ ei,
                            float* __restrict__ logits_p,   // in: logits, out: p
                            const unsigned* __restrict__ nmax,
                            float* __restrict__ denom) {
  int t = blockIdx.x * blockDim.x + threadIdx.x;
  if (t >= GAT_E * 4) return;
  const int e = t >> 2, h = t & 3;
  const int d = ei[GAT_E + e];
  const float mx = dec_f32(nmax[d * 4 + h]);
  const float p = __expf(logits_p[t] - mx);
  logits_p[t] = p;
  atomicAdd(&denom[d * 4 + h], p);
}

// ---------------------------------------------------------------------------
// Pass 3: alpha = p/denom[dst]; agg[dst] += alpha * xl[src]  (wave per edge)
// ---------------------------------------------------------------------------
__global__ void edge_aggregate(const int* __restrict__ ei,
                               const float* __restrict__ p,
                               const float* __restrict__ denom,
                               const float* __restrict__ xl,
                               float* __restrict__ agg,
                               float* __restrict__ alpha_out) {
  const int lane = threadIdx.x & 31;
  const int e = blockIdx.x * (blockDim.x >> 5) + (threadIdx.x >> 5);
  if (e >= GAT_E) return;
  const int s = ei[e];
  const int d = ei[GAT_E + e];
  float al[4];
#pragma unroll
  for (int h = 0; h < 4; ++h)
    al[h] = p[e * 4 + h] / (denom[d * 4 + h] + 1e-16f);
  if (alpha_out != nullptr && lane < 4) alpha_out[e * 4 + lane] = al[lane];

  const float* __restrict__ xls = xl + (size_t)s * 256;
  float* __restrict__ aggd = agg + (size_t)d * 256;
#pragma unroll
  for (int it = 0; it < 8; ++it) {
    const int c = it * 32 + lane;
    atomicAdd(&aggd[c], al[it >> 1] * xls[c]);
  }
}

// ---------------------------------------------------------------------------
// Node epilogue: h = silu(LayerNorm(agg + bias)) + residual
// ---------------------------------------------------------------------------
__global__ void node_post(const float* __restrict__ agg,
                          const float* __restrict__ bias,
                          const float* __restrict__ g,
                          const float* __restrict__ bb,
                          const float* __restrict__ residual,
                          float* __restrict__ hout,       // may alias residual / null
                          __bf16* __restrict__ hbout) {
  __shared__ float ssum[8], ssq[8];
  const int node = blockIdx.x;
  const int c = threadIdx.x;                    // 0..255
  const size_t idx = (size_t)node * 256 + c;
  const float v = agg[idx] + bias[c];

  float s = v, q = v * v;
#pragma unroll
  for (int off = 16; off > 0; off >>= 1) {
    s += __shfl_xor(s, off, 32);
    q += __shfl_xor(q, off, 32);
  }
  const int lane = c & 31, wv = c >> 5;
  if (lane == 0) { ssum[wv] = s; ssq[wv] = q; }
  __syncthreads();
  float tot = 0.f, totq = 0.f;
#pragma unroll
  for (int i = 0; i < 8; ++i) { tot += ssum[i]; totq += ssq[i]; }

  const float mu  = tot  * (1.f / 256.f);
  const float var = totq * (1.f / 256.f) - mu * mu;
  const float y   = (v - mu) * rsqrtf(var + 1e-5f) * g[c] + bb[c];
  const float sil = y / (1.f + __expf(-y));     // silu
  const float h   = sil + residual[idx];
  if (hout != nullptr) hout[idx] = h;
  hbout[idx] = (__bf16)h;
}

// ---------------------------------------------------------------------------
// Host-side orchestration (graph-capture safe: only async launches on stream)
// ---------------------------------------------------------------------------
extern "C" void kernel_launch(void* const* d_in, const int* in_sizes, int n_in,
                              void* d_out, int out_size, void* d_ws, size_t ws_size,
                              hipStream_t stream) {
  (void)in_sizes; (void)n_in; (void)out_size; (void)ws_size;
  constexpr int N = GAT_N, E = GAT_E, IN = GAT_IN, HID = GAT_HID;

  // ---- inputs (setup_inputs() dict order) ----
  const float* x      = (const float*)d_in[0];
  const int*   ei     = (const int*)  d_in[1];   // [2,E] int32
  const float* eattr  = (const float*)d_in[2];
  const float* w_res  = (const float*)d_in[3];
  const float* b_res  = (const float*)d_in[4];
  const float* w_l1   = (const float*)d_in[5];
  const float* b_l1   = (const float*)d_in[6];
  const float* w_r1   = (const float*)d_in[7];
  const float* b_r1   = (const float*)d_in[8];
  const float* w_e1   = (const float*)d_in[9];
  const float* att1   = (const float*)d_in[10];
  const float* bias1  = (const float*)d_in[11];
  const float* g1     = (const float*)d_in[12];
  const float* bb1    = (const float*)d_in[13];
  const float* w_l2   = (const float*)d_in[14];
  const float* b_l2   = (const float*)d_in[15];
  const float* w_r2   = (const float*)d_in[16];
  const float* b_r2   = (const float*)d_in[17];
  const float* w_e2   = (const float*)d_in[18];
  const float* att2   = (const float*)d_in[19];
  const float* bias2  = (const float*)d_in[20];
  const float* g2     = (const float*)d_in[21];
  const float* bb2    = (const float*)d_in[22];
  const float* w_p    = (const float*)d_in[23];
  const float* b_p    = (const float*)d_in[24];

  float* out_nodes = (float*)d_out;                       // [N,256]
  float* out_alpha = (float*)d_out + (size_t)N * 256;     // [E,4]

  // ---- workspace carve (aligned) ----
  char* ws = (char*)d_ws;
  size_t off = 0;
  auto carve = [&](size_t bytes) -> void* {
    off = (off + 255) & ~size_t(255);
    void* p = ws + off;
    off += bytes;
    return p;
  };
  float*    identity = (float*)   carve((size_t)N * 256 * 4);  // residual / h
  float*    xl       = (float*)   carve((size_t)N * 256 * 4);
  float*    xr       = (float*)   carve((size_t)N * 256 * 4);
  float*    agg      = (float*)   carve((size_t)N * 256 * 4);
  float*    logits   = (float*)   carve((size_t)E * 4 * 4);    // reused as p
  float*    denom    = (float*)   carve((size_t)N * 4 * 4);
  unsigned* nmax     = (unsigned*)carve((size_t)N * 4 * 4);
  __bf16*   xb       = (__bf16*)  carve((size_t)N * IN * 2);
  __bf16*   hb       = (__bf16*)  carve((size_t)N * 256 * 2);
  __bf16*   wb       = (__bf16*)  carve((size_t)(3 * IN * HID + 3 * HID * HID) * 2);
  __bf16* wres_b = wb;
  __bf16* wl1_b  = wb + IN * HID;
  __bf16* wr1_b  = wb + 2 * IN * HID;
  __bf16* wl2_b  = wb + 3 * IN * HID;
  __bf16* wr2_b  = wl2_b + HID * HID;
  __bf16* wp_b   = wr2_b + HID * HID;

  auto packw = [&](const float* src, __bf16* dst, int K) {
    pack_w_bf16<<<(K * 256 + 255) / 256, 256, 0, stream>>>(src, dst, K);
  };
  auto gemm = [&](const __bf16* A, const __bf16* Wp, const float* bias,
                  float* C, int M, int K) {
    dim3 grid(4, (M / 16 + 7) / 8);
    wmma_gemm_bf16<<<grid, 256, 0, stream>>>(A, Wp, bias, C, M, K);
  };

  const int edgeBlocks = (E + 7) / 8;                // wave-per-edge, 8 waves/block
  const int ehBlocks   = (E * 4 + 255) / 256;
  const int initBlocks = (N * 256 + 255) / 256;

  // ---- bf16 conversions / weight packing ----
  cvt_f32_bf16<<<(N * IN + 255) / 256, 256, 0, stream>>>(x, xb, N * IN);
  packw(w_res, wres_b, IN);
  packw(w_l1, wl1_b, IN);
  packw(w_r1, wr1_b, IN);
  packw(w_l2, wl2_b, HID);
  packw(w_r2, wr2_b, HID);
  packw(w_p,  wp_b,  HID);

  // ---- layer 1 dense transforms (WMMA) ----
  gemm(xb, wres_b, b_res, identity, N, IN);
  gemm(xb, wl1_b,  b_l1,  xl,       N, IN);
  gemm(xb, wr1_b,  b_r1,  xr,       N, IN);

  // ---- layer 1 edge softmax + aggregation ----
  init_layer<<<initBlocks, 256, 0, stream>>>(agg, denom, nmax);
  edge_logits<<<edgeBlocks, 256, 0, stream>>>(ei, eattr, xl, xr, w_e1, att1, logits, nmax);
  edge_expsum<<<ehBlocks, 256, 0, stream>>>(ei, logits, nmax, denom);
  edge_aggregate<<<edgeBlocks, 256, 0, stream>>>(ei, logits, denom, xl, agg, nullptr);
  node_post<<<N, 256, 0, stream>>>(agg, bias1, g1, bb1, identity, identity, hb);

  // ---- layer 2 dense transforms (WMMA, K=256) ----
  gemm(hb, wl2_b, b_l2, xl, N, HID);
  gemm(hb, wr2_b, b_r2, xr, N, HID);

  // ---- layer 2 edge softmax + aggregation (alpha -> d_out) ----
  init_layer<<<initBlocks, 256, 0, stream>>>(agg, denom, nmax);
  edge_logits<<<edgeBlocks, 256, 0, stream>>>(ei, eattr, xl, xr, w_e2, att2, logits, nmax);
  edge_expsum<<<ehBlocks, 256, 0, stream>>>(ei, logits, nmax, denom);
  edge_aggregate<<<edgeBlocks, 256, 0, stream>>>(ei, logits, denom, xl, agg, out_alpha);
  node_post<<<N, 256, 0, stream>>>(agg, bias2, g2, bb2, identity, nullptr, hb);

  // ---- projector (WMMA) straight into d_out ----
  gemm(hb, wp_b, b_p, out_nodes, N, HID);
}
